// NaiveCausalSelfAttention_13271448945324
// MI455X (gfx1250) — matrix operand order repaired and checked by reading, not compiled, not run on hardware
//
#include <hip/hip_runtime.h>
#include <hip/hip_bf16.h>

// ---------------------------------------------------------------------------
// Problem constants (from reference)
// ---------------------------------------------------------------------------
#define S_LEN 2048
#define HDIM  2048
#define NH    32
#define NKV   8
#define HD    64
#define LDQ   (NH * HD)    // 2048
#define LDKV  (NKV * HD)   // 512

typedef unsigned int  uint_t;
typedef unsigned short ushort_t;

typedef __attribute__((ext_vector_type(16))) __bf16        v16bf;
typedef __attribute__((ext_vector_type(8)))  float         v8f;
typedef __attribute__((ext_vector_type(8)))  unsigned int  v8u;

// ---------------------------------------------------------------------------
// Helpers
// ---------------------------------------------------------------------------
__device__ __forceinline__ ushort_t f2bf(float f) {
    unsigned int u = __float_as_uint(f);
    unsigned int r = u + 0x7FFFu + ((u >> 16) & 1u);   // round-to-nearest-even
    return (ushort_t)(r >> 16);
}

// gfx1250 async copy global -> LDS (ASYNCcnt-tracked), via inline asm so it is
// portable across ROCm / upstream toolchains (builtin arity differs).
__device__ __forceinline__ void async_ld_b128(const ushort_t* g, ushort_t* l) {
    unsigned lds_addr = (unsigned)(uintptr_t)l;              // low 32 bits = LDS byte addr
    unsigned long long ga = (unsigned long long)(uintptr_t)g;
    asm volatile("global_load_async_to_lds_b128 %0, %1, off"
                 :: "v"(lds_addr), "v"(ga)
                 : "memory");
}
__device__ __forceinline__ void wait_async0() {
    asm volatile("s_wait_asynccnt 0x0" ::: "memory");
}

// A-matrix fragment (16x32 bf16, CDNA5 ISA 16-bit A layout), global source.
__device__ __forceinline__ v16bf load_afrag(const ushort_t* A, int ldk,
                                            int row, int k0, int lane) {
    const ushort_t* base = A + (size_t)row * ldk + k0;
    const int half8 = (lane & 16) ? 8 : 0;
    v8u d;
#pragma unroll
    for (int j = 0; j < 8; ++j) {
        int off = ((j < 4) ? 0 : 16) + half8 + 2 * (j & 3);
        d[j] = *reinterpret_cast<const uint_t*>(base + off);
    }
    return __builtin_bit_cast(v16bf, d);
}

// B-matrix fragment (32x16 bf16): lane holds column 'col'; K in A-table order (8h+e, 16+8h+e-8).
// Source stored transposed: Bt[col][k] contiguous in k.
__device__ __forceinline__ v16bf load_bfrag(const ushort_t* Bt, int ldk,
                                            int col, int k0, int lane) {
    const ushort_t* base = Bt + (size_t)col * ldk + k0 + ((lane & 16) ? 8 : 0);
    v8u d;
#pragma unroll
    for (int j = 0; j < 8; ++j)
        d[j] = *reinterpret_cast<const uint_t*>(base + ((j < 4) ? 0 : 16) + 2 * (j & 3));
    return __builtin_bit_cast(v16bf, d);
}

// Fragment from LDS: two 16B chunks at p and p+second_off (ushort units).
// A and B fragments: second_off = 16 (8-element gap).
__device__ __forceinline__ v16bf frag_from_lds(const ushort_t* p, int second_off) {
    uint4 a = *reinterpret_cast<const uint4*>(p);
    uint4 b = *reinterpret_cast<const uint4*>(p + second_off);
    v8u d;
    d[0] = a.x; d[1] = a.y; d[2] = a.z; d[3] = a.w;
    d[4] = b.x; d[5] = b.y; d[6] = b.z; d[7] = b.w;
    return __builtin_bit_cast(v16bf, d);
}

// ---------------------------------------------------------------------------
// Conversion kernels
// ---------------------------------------------------------------------------
__global__ void k_f32_to_bf16(const float* __restrict__ in,
                              ushort_t* __restrict__ out, int n) {
    int i = blockIdx.x * blockDim.x + threadIdx.x;
    if (i < n) out[i] = f2bf(in[i]);
}

// W[K][N] (row-major f32) -> Wt[N][K] (bf16)
__global__ void k_transpose_bf16(const float* __restrict__ W,
                                 ushort_t* __restrict__ Wt, int K, int N) {
    int i = blockIdx.x * blockDim.x + threadIdx.x;
    if (i < K * N) {
        int k = i / N, n = i % N;
        Wt[(size_t)n * K + k] = f2bf(W[i]);
    }
}

// RoPE on f32 [S][nh][64] -> bf16 same layout. One thread per (s, h, d<32).
__global__ void k_rope_bf16(const float* __restrict__ src,
                            const int* __restrict__ pos,
                            ushort_t* __restrict__ dst, int nheads) {
    int i = blockIdx.x * blockDim.x + threadIdx.x;
    int total = S_LEN * nheads * 32;
    if (i >= total) return;
    int d = i & 31;
    int h = (i >> 5) % nheads;
    int s = i / (32 * nheads);
    float p = (float)pos[s];
    float inv_freq = __powf(10000.0f, -(2.0f * (float)d) / 64.0f);
    float sn, cs;
    __sincosf(p * inv_freq, &sn, &cs);
    size_t base = (size_t)s * nheads * 64 + (size_t)h * 64 + d;
    float x1 = src[base], x2 = src[base + 32];
    dst[base]      = f2bf(x1 * cs - x2 * sn);
    dst[base + 32] = f2bf(x2 * cs + x1 * sn);
}

// V f32 [s][kvh*64+d] -> bf16 VT[(kvh*64+d)][s]
__global__ void k_v_transpose_bf16(const float* __restrict__ vf,
                                   ushort_t* __restrict__ vt) {
    int i = blockIdx.x * blockDim.x + threadIdx.x;
    if (i < S_LEN * LDKV) {
        int c = i % LDKV, s = i / LDKV;
        vt[(size_t)c * S_LEN + s] = f2bf(vf[i]);
    }
}

// ---------------------------------------------------------------------------
// WMMA GEMM with double-buffered async-to-LDS staging.
// C[M][N] (f32) = A[M][K] (bf16 row-major) @ Bt[N][K]^T (bf16).
// Block = 128 threads (4 waves) computes a 64x64 tile: wave w owns rows
// [w*16, w*16+16) and all 64 cols; A/B 64x32 k-tiles staged in LDS.
// Grid = (M/64, N/64).
// ---------------------------------------------------------------------------
__global__ __launch_bounds__(128)
void k_gemm_bf16_wmma(const ushort_t* __restrict__ A,
                      const ushort_t* __restrict__ Bt,
                      float* __restrict__ C, int M, int N, int K) {
    __shared__ ushort_t As[2][64][40];    // 80B row stride: conflict-free-ish
    __shared__ ushort_t Bs[2][64][40];
    const int lane = threadIdx.x & 31;
    const int wave = threadIdx.x >> 5;
    const int rowB = blockIdx.x * 64;
    const int colB = blockIdx.y * 64;
    const int half8  = (lane & 16) ? 8 : 0;
    const int half16 = (lane & 16) ? 16 : 0;
    const int l15 = lane & 15;

    v8f acc[4];
#pragma unroll
    for (int t = 0; t < 4; ++t) acc[t] = (v8f)0.0f;

    // stage one 64x32 k-tile of A and of B into LDS buffer `buf`
    auto stage = [&](int k0, int buf) {
#pragma unroll
        for (int i = 0; i < 2; ++i) {
            int c = (int)threadIdx.x + i * 128;   // 0..255 -> 64 rows x 4 segs
            int r = c >> 2, seg = (c & 3) * 8;
            async_ld_b128(A  + (size_t)(rowB + r) * K + k0 + seg, &As[buf][r][seg]);
            async_ld_b128(Bt + (size_t)(colB + r) * K + k0 + seg, &Bs[buf][r][seg]);
        }
    };

    stage(0, 0);
    wait_async0();
    __syncthreads();

    int buf = 0;
    for (int k0 = 0; k0 < K; k0 += 32) {
        if (k0 + 32 < K) stage(k0 + 32, buf ^ 1);   // overlap copy with WMMA

        v16bf a = frag_from_lds(&As[buf][wave * 16 + l15][half8], 16);
#pragma unroll
        for (int t = 0; t < 4; ++t) {
            v16bf b = frag_from_lds(&Bs[buf][t * 16 + l15][half8], 16);
            acc[t] = __builtin_amdgcn_wmma_f32_16x16x32_bf16(
                false, a, false, b, (short)0, acc[t], false, false);
        }

        wait_async0();       // next tile landed
        __syncthreads();     // all waves done reading current tile
        buf ^= 1;
    }

    const int rowOff = (lane & 16) ? 8 : 0;
    const int row0w = rowB + wave * 16;
#pragma unroll
    for (int t = 0; t < 4; ++t)
#pragma unroll
        for (int r = 0; r < 8; ++r)
            C[(size_t)(row0w + rowOff + r) * N + colB + t * 16 + l15] = acc[t][r];
}

// ---------------------------------------------------------------------------
// Flash-style causal attention. Grid = (S/16, NH), block = 32 (one wave).
// QB bf16 [S][NH*64] (RoPE applied), KB bf16 [S][NKV*64] (RoPE applied),
// VT bf16 [NKV*64][S], CTX f32 [S][NH*64].
// ---------------------------------------------------------------------------
__global__ __launch_bounds__(32)
void k_attention_wmma(const ushort_t* __restrict__ QB,
                      const ushort_t* __restrict__ KB,
                      const ushort_t* __restrict__ VT,
                      float* __restrict__ CTX) {
    const int lane = threadIdx.x & 31;
    const int qt = blockIdx.x;
    const int h  = blockIdx.y;
    const int hk = h >> 2;                   // NH/NKV = 4
    const int qs = qt * 16;
    const int rowOff = (lane & 16) ? 8 : 0;
    const int col = lane & 15;

    __shared__ ushort_t P[16][40];           // 16x32 probs, padded stride

    // Q fragments (A layout, K-dim = head_dim), loaded once.
    const v16bf qf0 = load_afrag(QB + h * 64, LDQ, qs + col, 0, lane);
    const v16bf qf1 = load_afrag(QB + h * 64, LDQ, qs + col, 32, lane);

    v8f o[4];
#pragma unroll
    for (int t = 0; t < 4; ++t) o[t] = (v8f)0.0f;
    float m[8], l[8];
#pragma unroll
    for (int r = 0; r < 8; ++r) { m[r] = -3.0e38f; l[r] = 0.0f; }

    const int nblk = (qs + 15) / 32 + 1;     // key blocks of 32, causal bound
    for (int b = 0; b < nblk; ++b) {
        const int kb = b * 32;
        if (b + 1 < nblk)                    // gfx1250 global_prefetch_b8
            __builtin_prefetch(KB + (size_t)(kb + 32 + col) * LDKV + hk * 64, 0, 1);

        // ---- scores: S = Q @ K^T  (two 16x16 tiles over 32 keys) ----
        v16bf k00 = load_bfrag(KB + hk * 64, LDKV, kb + col,      0,  lane);
        v16bf k01 = load_bfrag(KB + hk * 64, LDKV, kb + col,      32, lane);
        v16bf k10 = load_bfrag(KB + hk * 64, LDKV, kb + 16 + col, 0,  lane);
        v16bf k11 = load_bfrag(KB + hk * 64, LDKV, kb + 16 + col, 32, lane);
        v8f s0 = (v8f)0.0f, s1 = (v8f)0.0f;
        s0 = __builtin_amdgcn_wmma_f32_16x16x32_bf16(false, qf0, false, k00,
                                                     (short)0, s0, false, false);
        s0 = __builtin_amdgcn_wmma_f32_16x16x32_bf16(false, qf1, false, k01,
                                                     (short)0, s0, false, false);
        s1 = __builtin_amdgcn_wmma_f32_16x16x32_bf16(false, qf0, false, k10,
                                                     (short)0, s1, false, false);
        s1 = __builtin_amdgcn_wmma_f32_16x16x32_bf16(false, qf1, false, k11,
                                                     (short)0, s1, false, false);

        // ---- scale + causal mask + online softmax ----
#pragma unroll
        for (int r = 0; r < 8; ++r) {
            const int q = qs + rowOff + r;
            float v0 = s0[r] * 0.125f;
            float v1 = s1[r] * 0.125f;
            if (kb + col > q)      v0 = -3.0e38f;
            if (kb + 16 + col > q) v1 = -3.0e38f;

            float mx = fmaxf(v0, v1);
            mx = fmaxf(mx, __shfl_xor(mx, 1, 32));
            mx = fmaxf(mx, __shfl_xor(mx, 2, 32));
            mx = fmaxf(mx, __shfl_xor(mx, 4, 32));
            mx = fmaxf(mx, __shfl_xor(mx, 8, 32));

            const float mn = fmaxf(m[r], mx);
            const float sc = __expf(m[r] - mn);
            const float p0 = __expf(v0 - mn);
            const float p1 = __expf(v1 - mn);
            float sum = p0 + p1;
            sum += __shfl_xor(sum, 1, 32);
            sum += __shfl_xor(sum, 2, 32);
            sum += __shfl_xor(sum, 4, 32);
            sum += __shfl_xor(sum, 8, 32);

            l[r] = l[r] * sc + sum;
            m[r] = mn;
#pragma unroll
            for (int t = 0; t < 4; ++t) o[t][r] *= sc;

            P[rowOff + r][col]      = f2bf(p0);
            P[rowOff + r][16 + col] = f2bf(p1);
        }
        __syncthreads();

        // ---- P (C layout in LDS) -> A fragment ----
        {
            const ushort_t* prow = &P[lane & 15][0];
            const int half8 = (lane & 16) ? 8 : 0;
            v8u pd;
#pragma unroll
            for (int j = 0; j < 8; ++j) {
                int off = ((j < 4) ? 0 : 16) + half8 + 2 * (j & 3);
                pd[j] = *reinterpret_cast<const uint_t*>(prow + off);
            }
            v16bf pf = __builtin_bit_cast(v16bf, pd);

            // ---- O += P @ V ----
#pragma unroll
            for (int t = 0; t < 4; ++t) {
                v16bf vf = load_bfrag(VT + (size_t)(hk * 64) * S_LEN, S_LEN,
                                      t * 16 + col, kb, lane);
                o[t] = __builtin_amdgcn_wmma_f32_16x16x32_bf16(
                    false, pf, false, vf, (short)0, o[t], false, false);
            }
        }
        __syncthreads();
    }

    // ---- epilogue: normalize and store context ----
#pragma unroll
    for (int r = 0; r < 8; ++r) {
        const float invl = 1.0f / l[r];
#pragma unroll
        for (int t = 0; t < 4; ++t)
            CTX[(size_t)(qs + rowOff + r) * LDQ + h * 64 + t * 16 + col] =
                o[t][r] * invl;
    }
}

// ---------------------------------------------------------------------------
// Host-side launcher
// ---------------------------------------------------------------------------
extern "C" void kernel_launch(void* const* d_in, const int* in_sizes, int n_in,
                              void* d_out, int out_size, void* d_ws, size_t ws_size,
                              hipStream_t stream) {
    (void)in_sizes; (void)n_in; (void)out_size; (void)ws_size;

    const float* x   = (const float*)d_in[0];
    const int*   pos = (const int*)d_in[1];
    const float* wq  = (const float*)d_in[2];
    const float* wk  = (const float*)d_in[3];
    const float* wv  = (const float*)d_in[4];
    const float* wo  = (const float*)d_in[5];
    float* out = (float*)d_out;

    char* ws = (char*)d_ws;
    const size_t MB = 1ull << 20;
    ushort_t* XB  = (ushort_t*)(ws + 0 * MB);    // 8 MB  x bf16
    ushort_t* WQT = (ushort_t*)(ws + 8 * MB);    // 8 MB
    ushort_t* WKT = (ushort_t*)(ws + 16 * MB);   // 2 MB
    ushort_t* WVT = (ushort_t*)(ws + 18 * MB);   // 2 MB
    ushort_t* WOT = (ushort_t*)(ws + 20 * MB);   // 8 MB
    float*    QF  = (float*)(ws + 28 * MB);      // 16 MB  (later: CTX f32)
    float*    KF  = (float*)(ws + 44 * MB);      // 4 MB   (later: CTX bf16 lo)
    float*    VF  = (float*)(ws + 48 * MB);      // 4 MB   (later: CTX bf16 hi)
    ushort_t* QB  = (ushort_t*)(ws + 52 * MB);   // 8 MB
    ushort_t* KB  = (ushort_t*)(ws + 60 * MB);   // 2 MB
    ushort_t* VT  = (ushort_t*)(ws + 62 * MB);   // 2 MB
    float*    CTXF = QF;                          // alias (QF dead after RoPE)
    ushort_t* CTXB = (ushort_t*)KF;               // alias (KF/VF dead after RoPE)

    const int T = 256;
    // 1) conversions
    k_f32_to_bf16<<<(S_LEN * HDIM + T - 1) / T, T, 0, stream>>>(x, XB, S_LEN * HDIM);
    k_transpose_bf16<<<(HDIM * LDQ  + T - 1) / T, T, 0, stream>>>(wq, WQT, HDIM, LDQ);
    k_transpose_bf16<<<(HDIM * LDKV + T - 1) / T, T, 0, stream>>>(wk, WKT, HDIM, LDKV);
    k_transpose_bf16<<<(HDIM * LDKV + T - 1) / T, T, 0, stream>>>(wv, WVT, HDIM, LDKV);
    k_transpose_bf16<<<(LDQ  * HDIM + T - 1) / T, T, 0, stream>>>(wo, WOT, LDQ, HDIM);

    // 2) projections (WMMA GEMMs, async-LDS double buffered)
    k_gemm_bf16_wmma<<<dim3(S_LEN / 64, LDQ  / 64), 128, 0, stream>>>(XB, WQT, QF, S_LEN, LDQ,  HDIM);
    k_gemm_bf16_wmma<<<dim3(S_LEN / 64, LDKV / 64), 128, 0, stream>>>(XB, WKT, KF, S_LEN, LDKV, HDIM);
    k_gemm_bf16_wmma<<<dim3(S_LEN / 64, LDKV / 64), 128, 0, stream>>>(XB, WVT, VF, S_LEN, LDKV, HDIM);

    // 3) RoPE + layout conversions
    k_rope_bf16<<<(S_LEN * NH  * 32 + T - 1) / T, T, 0, stream>>>(QF, pos, QB, NH);
    k_rope_bf16<<<(S_LEN * NKV * 32 + T - 1) / T, T, 0, stream>>>(KF, pos, KB, NKV);
    k_v_transpose_bf16<<<(S_LEN * LDKV + T - 1) / T, T, 0, stream>>>(VF, VT);

    // 4) attention (WMMA flash)
    k_attention_wmma<<<dim3(S_LEN / 16, NH), 32, 0, stream>>>(QB, KB, VT, CTXF);

    // 5) output projection
    k_f32_to_bf16<<<(S_LEN * LDQ + T - 1) / T, T, 0, stream>>>(CTXF, CTXB, S_LEN * LDQ);
    k_gemm_bf16_wmma<<<dim3(S_LEN / 16 / 4, HDIM / 64), 128, 0, stream>>>(CTXB, WOT, out, S_LEN, HDIM, HDIM);
}
